// Decoder_56092272886279
// MI455X (gfx1250) — compile-verified, matrix operand order checked by reading, not run on hardware
//
#include <hip/hip_runtime.h>
#include <math.h>

typedef float  v2f   __attribute__((ext_vector_type(2)));
typedef float  v8f   __attribute__((ext_vector_type(8)));
typedef __bf16 v8bf  __attribute__((ext_vector_type(8)));
typedef __bf16 v16bf __attribute__((ext_vector_type(16)));

#define T_LEN   256
#define B_SZ    512
#define Q_DIM   64
#define H1_D    128
#define H2_D    64
#define EMB_D   128
#define LTEXT   30
#define L_STEPS 29
#define VOCAB_N 5000
#define K1      320      // EMB + Q + H1   (x | ctx | h1)
#define N1      512      // 4*H1
#define K2      192      // H1 + H2
#define N2      256      // 4*H2
#define KS1     (K1/4)   // 80 f32-WMMA k-steps
#define KS2     (K2/4)   // 48
#define KSQ     (Q_DIM/4)// 16
#define NT1     (N1/16)  // 32
#define NT2     (N2/16)  // 16
#define NTQ     (Q_DIM/16) // 4
#define WO_LDN  5120     // padded N for output GEMM (40 blocks * 128)
#define WO_NT   (WO_LDN/16)   // 320 n-tiles
#define WO_KB   (EMB_D/32)    // 4 k-blocks of 32

// ---- workspace layout (units: float) ----
#define OFF_HC  0
#define SZ_HC   (L_STEPS * B_SZ * (H2_D + Q_DIM))     // element count (stored as bf16)
#define OFF_W1  (OFF_HC + SZ_HC)
#define SZ_W1   (K1 * N1)                              // 163,840 (f32 B-fragment order)
#define OFF_W2  (OFF_W1 + SZ_W1)
#define SZ_W2   (K2 * N2)                              // 49,152
#define OFF_WQ  (OFF_W2 + SZ_W2)
#define SZ_WQ   (H2_D * Q_DIM)                         // 4,096
#define OFF_B1  (OFF_WQ + SZ_WQ)
#define OFF_B2  (OFF_B1 + N1)
#define OFF_WO  (OFF_B2 + N2)                          // bf16 fragments live here
#define SZ_WO_BF16 (WO_NT * WO_KB * 32 * 16)           // 655,360 bf16

__device__ __forceinline__ float sigf(float x) { return 1.0f / (1.0f + __expf(-x)); }

// ---------------------------------------------------------------------------
// Kernel 0: repack weights.
//  * W1/W2/Wq -> fp32 per-lane B-fragment order for V_WMMA_F32_16X16X4_F32:
//      frag[(((nt*KS)+ks)*32+lane)*2+j] = W[k=ks*4+(lane>>4)*2+j][n=nt*16+(lane&15)]
//    so each WMMA consumes one coalesced 8-byte load per lane.
//  * W_out -> bf16 per-lane B-fragment order for V_WMMA_F32_16X16X32_BF16.
//  * bias pairs pre-summed.
// ---------------------------------------------------------------------------
__global__ void dec_pack_kernel(const float* __restrict__ Wih1, const float* __restrict__ Whh1,
                                const float* __restrict__ bih1, const float* __restrict__ bhh1,
                                const float* __restrict__ Wih2, const float* __restrict__ Whh2,
                                const float* __restrict__ bih2, const float* __restrict__ bhh2,
                                const float* __restrict__ Wq,   const float* __restrict__ Wout,
                                float* __restrict__ ws) {
  int idx    = blockIdx.x * blockDim.x + threadIdx.x;
  int stride = gridDim.x * blockDim.x;

  for (int i = idx; i < SZ_W1; i += stride) {
    int j = i & 1, lane = (i >> 1) & 31, rest = i >> 6;
    int ks = rest % KS1, nt = rest / KS1;
    int k = ks * 4 + (lane >> 4) * 2 + j;
    int n = nt * 16 + (lane & 15);
    ws[OFF_W1 + i] = (k < 192) ? Wih1[n * 192 + k] : Whh1[n * 128 + (k - 192)];
  }
  for (int i = idx; i < SZ_W2; i += stride) {
    int j = i & 1, lane = (i >> 1) & 31, rest = i >> 6;
    int ks = rest % KS2, nt = rest / KS2;
    int k = ks * 4 + (lane >> 4) * 2 + j;
    int n = nt * 16 + (lane & 15);
    ws[OFF_W2 + i] = (k < 128) ? Wih2[n * 128 + k] : Whh2[n * 64 + (k - 128)];
  }
  for (int i = idx; i < SZ_WQ; i += stride) {
    int j = i & 1, lane = (i >> 1) & 31, rest = i >> 6;
    int ks = rest % KSQ, nt = rest / KSQ;
    int k = ks * 4 + (lane >> 4) * 2 + j;
    int n = nt * 16 + (lane & 15);
    ws[OFF_WQ + i] = Wq[n * H2_D + k];
  }
  for (int i = idx; i < N1; i += stride) ws[OFF_B1 + i] = bih1[i] + bhh1[i];
  for (int i = idx; i < N2; i += stride) ws[OFF_B2 + i] = bih2[i] + bhh2[i];

  __bf16* wofrag = (__bf16*)(ws + OFF_WO);
  for (int i = idx; i < SZ_WO_BF16; i += stride) {
    int e    = i & 15;
    int lane = (i >> 4) & 31;
    int kb   = (i >> 9) & 3;
    int nt   = i >> 11;
    int n    = nt * 16 + (lane & 15);
    int half = lane >> 4;
    int v    = e >> 1;
    int kin  = ((v < 4) ? (2 * v) : (16 + 2 * (v - 4))) + half * 8 + (e & 1);
    int k    = kb * 32 + kin;
    float val = (n < VOCAB_N) ? Wout[n * EMB_D + k] : 0.0f;
    wofrag[i] = (__bf16)val;
  }
}

// ---------------------------------------------------------------------------
// Kernel 1: recurrent decoder. One block = 16 batch rows for all 29 steps.
// LSTM GEMMs use fp32 WMMA (16x16x4). Writes hc[t][b][0:128] = h2 || ctx (bf16).
// ---------------------------------------------------------------------------
__global__ __launch_bounds__(256) void dec_recurrent_kernel(
    const float* __restrict__ key, const float* __restrict__ values,
    const float* __restrict__ mask, const int* __restrict__ text,
    const float* __restrict__ emb,  const float* __restrict__ bq,
    const float* __restrict__ w1f,  const float* __restrict__ w2f,
    const float* __restrict__ wqf,  const float* __restrict__ b1,
    const float* __restrict__ b2,   __bf16* __restrict__ hcbf) {
  __shared__ float X1[16 * K1];     // [ x(128) | ctx(64) | h1(128) ] per row
  __shared__ float G[16 * N1];      // reused: g1 (512) / g2 (256) / energy (256)
  __shared__ float H2[16 * H2_D];
  __shared__ float QS[16 * Q_DIM];
  __shared__ float red[256];
  __shared__ float rowmax[16];
  __shared__ float rowinv[16];

  const int tid  = threadIdx.x;
  const int lane = tid & 31;
  const int wave = tid >> 5;
  const int half = lane >> 4;       // 0|1
  const int mr   = lane & 15;       // M (A) / N (B) index within tile
  const int koff = half * 2;        // f32 WMMA K sub-offset
  const int b0   = blockIdx.x * 16;

  const int gm   = tid >> 4;        // gate row 0..15
  const int gnb1 = (tid & 15) * 8;  // LSTM1: 8 gate cols / thread
  const int gnb2 = (tid & 15) * 4;  // LSTM2: 4 gate cols / thread
  float c1r[8], c2r[4];
#pragma unroll
  for (int j = 0; j < 8; ++j) c1r[j] = 0.f;
#pragma unroll
  for (int j = 0; j < 4; ++j) c2r[j] = 0.f;

  for (int i = tid; i < 16 * K1; i += 256) X1[i] = 0.f;       // x/ctx/h1 = 0
  for (int i = tid; i < 16 * H2_D; i += 256) H2[i] = 0.f;
  __syncthreads();

  for (int t = 0; t < L_STEPS; ++t) {
    // --- embedding gather into X1[:, 0:128] ---
    {
      int m  = tid >> 4;
      int kb = (tid & 15) * 8;
      int tok = text[(b0 + m) * LTEXT + t];
      const float* er = emb + (size_t)tok * EMB_D + kb;
#pragma unroll
      for (int j = 0; j < 8; ++j) X1[m * K1 + kb + j] = er[j];
    }
    __syncthreads();

    // --- GEMM1: G[16][512] = X1(16x320) @ W1(320x512), fp32 WMMA ---
    for (int nt = wave; nt < NT1; nt += 8) {
      int n0 = nt * 16;
      const float* bf = w1f + (size_t)nt * KS1 * 64 + lane * 2;
      v8f acc = {0.f, 0.f, 0.f, 0.f, 0.f, 0.f, 0.f, 0.f};
      for (int ks = 0; ks < KS1; ++ks) {
        v2f a, b;
        const float* ap = &X1[mr * K1 + ks * 4 + koff];
        a.x = ap[0]; a.y = ap[1];
        b = *(const v2f*)(bf + ks * 64);
        acc = __builtin_amdgcn_wmma_f32_16x16x4_f32(false, a, false, b, (short)0, acc, false, false);
      }
#pragma unroll
      for (int r = 0; r < 8; ++r) G[(r + half * 8) * N1 + n0 + mr] = acc[r];
    }
    __syncthreads();

    // --- LSTM1 gates: c1 in registers, h1 -> X1[:, 192:320] ---
#pragma unroll
    for (int j = 0; j < 8; ++j) {
      int n = gnb1 + j;
      float ig = G[gm * N1 + n]        + b1[n];
      float fg = G[gm * N1 + 128 + n]  + b1[128 + n];
      float gg = G[gm * N1 + 256 + n]  + b1[256 + n];
      float og = G[gm * N1 + 384 + n]  + b1[384 + n];
      float c  = sigf(fg) * c1r[j] + sigf(ig) * tanhf(gg);
      c1r[j] = c;
      X1[gm * K1 + 192 + n] = sigf(og) * tanhf(c);
    }
    __syncthreads();

    // --- GEMM2: G[16][256] = [h1|h2](16x192) @ W2(192x256), fp32 WMMA ---
    for (int nt = wave; nt < NT2; nt += 8) {
      int n0 = nt * 16;
      const float* bf = w2f + (size_t)nt * KS2 * 64 + lane * 2;
      v8f acc = {0.f, 0.f, 0.f, 0.f, 0.f, 0.f, 0.f, 0.f};
      for (int ks = 0; ks < KS2; ++ks) {
        v2f a, b;
        int ka = ks * 4 + koff;
        if (ka < 128) { a.x = X1[mr * K1 + 192 + ka]; a.y = X1[mr * K1 + 192 + ka + 1]; }
        else          { a.x = H2[mr * H2_D + (ka - 128)]; a.y = H2[mr * H2_D + (ka - 127)]; }
        b = *(const v2f*)(bf + ks * 64);
        acc = __builtin_amdgcn_wmma_f32_16x16x4_f32(false, a, false, b, (short)0, acc, false, false);
      }
#pragma unroll
      for (int r = 0; r < 8; ++r) G[(r + half * 8) * N2 + n0 + mr] = acc[r];
    }
    __syncthreads();

    // --- LSTM2 gates: c2 in registers, new h2 -> H2 ---
#pragma unroll
    for (int j = 0; j < 4; ++j) {
      int n = gnb2 + j;
      float ig = G[gm * N2 + n]        + b2[n];
      float fg = G[gm * N2 + 64 + n]   + b2[64 + n];
      float gg = G[gm * N2 + 128 + n]  + b2[128 + n];
      float og = G[gm * N2 + 192 + n]  + b2[192 + n];
      float c  = sigf(fg) * c2r[j] + sigf(ig) * tanhf(gg);
      c2r[j] = c;
      H2[gm * H2_D + n] = sigf(og) * tanhf(c);
    }
    __syncthreads();

    // --- q = h2 @ WqT (16x64)@(64x64), fp32 WMMA on waves 0..3 ---
    if (wave < 4) {
      int n0 = wave * 16;
      const float* bf = wqf + (size_t)wave * KSQ * 64 + lane * 2;
      v8f acc = {0.f, 0.f, 0.f, 0.f, 0.f, 0.f, 0.f, 0.f};
      for (int ks = 0; ks < KSQ; ++ks) {
        v2f a, b;
        a.x = H2[mr * H2_D + ks * 4 + koff]; a.y = H2[mr * H2_D + ks * 4 + koff + 1];
        b = *(const v2f*)(bf + ks * 64);
        acc = __builtin_amdgcn_wmma_f32_16x16x4_f32(false, a, false, b, (short)0, acc, false, false);
      }
#pragma unroll
      for (int r = 0; r < 8; ++r) QS[(r + half * 8) * Q_DIM + n0 + mr] = acc[r];
    }
    __syncthreads();
    {   // + b_q
      int m = tid >> 4, nb = (tid & 15) * 4;
#pragma unroll
      for (int j = 0; j < 4; ++j) QS[m * Q_DIM + nb + j] += bq[nb + j];
    }
    __syncthreads();

    // --- attention energies: e[m][t] = q[m] . key[t][b0+m] ---
    {
      int m = tid >> 4, tc = tid & 15, b = b0 + m;
      float lmax = -3.4e38f;
      for (int tt = 0; tt < 16; ++tt) {
        int ti = tc * 16 + tt;
        const float* kr = key + ((size_t)ti * B_SZ + b) * Q_DIM;
        float acc = 0.f;
#pragma unroll
        for (int kk = 0; kk < Q_DIM; kk += 4) {
          acc += QS[m * Q_DIM + kk] * kr[kk] + QS[m * Q_DIM + kk + 1] * kr[kk + 1]
               + QS[m * Q_DIM + kk + 2] * kr[kk + 2] + QS[m * Q_DIM + kk + 3] * kr[kk + 3];
        }
        G[m * T_LEN + ti] = acc;
        lmax = fmaxf(lmax, acc);
      }
      red[tid] = lmax;
    }
    __syncthreads();
    if (tid < 16) {
      float mx = red[tid * 16];
      for (int j = 1; j < 16; ++j) mx = fmaxf(mx, red[tid * 16 + j]);
      rowmax[tid] = mx;
    }
    __syncthreads();
    {   // masked exp (softmax denom cancels after renorm)
      int m = tid >> 4, tc = tid & 15, b = b0 + m;
      float mx = rowmax[m], ls = 0.f;
      for (int tt = 0; tt < 16; ++tt) {
        int ti = tc * 16 + tt;
        float w = mask[b * T_LEN + ti] * __expf(G[m * T_LEN + ti] - mx);
        G[m * T_LEN + ti] = w;
        ls += w;
      }
      red[tid] = ls;
    }
    __syncthreads();
    if (tid < 16) {
      float s = 0.f;
      for (int j = 0; j < 16; ++j) s += red[tid * 16 + j];
      rowinv[tid] = 1.0f / fmaxf(s, 1e-30f);
    }
    __syncthreads();
    {   // ctx -> X1[:, 128:192]
      int m = tid >> 4, nb = (tid & 15) * 4, b = b0 + m;
      float inv = rowinv[m];
      float a0 = 0.f, a1 = 0.f, a2 = 0.f, a3 = 0.f;
      for (int ti = 0; ti < T_LEN; ++ti) {
        float w = G[m * T_LEN + ti];
        const float* vr = values + ((size_t)ti * B_SZ + b) * Q_DIM + nb;
        a0 += w * vr[0]; a1 += w * vr[1]; a2 += w * vr[2]; a3 += w * vr[3];
      }
      X1[m * K1 + 128 + nb + 0] = a0 * inv;
      X1[m * K1 + 128 + nb + 1] = a1 * inv;
      X1[m * K1 + 128 + nb + 2] = a2 * inv;
      X1[m * K1 + 128 + nb + 3] = a3 * inv;
    }
    __syncthreads();
    {   // hc[t][b][:] = h2 || ctx  (bf16, one 16B store per thread)
      int m = tid >> 4, kb = (tid & 15) * 8, b = b0 + m;
      v8bf vals;
#pragma unroll
      for (int j = 0; j < 8; ++j) {
        int k = kb + j;
        float v = (k < 64) ? H2[m * H2_D + k] : X1[m * K1 + 128 + (k - 64)];
        vals[j] = (__bf16)v;
      }
      *(v8bf*)(hcbf + ((size_t)t * B_SZ + b) * 128 + kb) = vals;
    }
    __syncthreads();
  }
}

// ---------------------------------------------------------------------------
// Kernel 2: output projection (29*512 x 5000, K=128) with bf16 WMMA.
// A tile staged LDS via GLOBAL_LOAD_ASYNC_TO_LDS_B128 (ASYNCcnt).
// grid (928, 40): x = 16-row M tile, y*8+wave = 16-col N tile.
// ---------------------------------------------------------------------------
__global__ __launch_bounds__(256) void dec_outgemm_kernel(
    const __bf16* __restrict__ hcbf, const __bf16* __restrict__ wofrag,
    const float* __restrict__ bout, float* __restrict__ out) {
  __shared__ __bf16 Abf[16 * EMB_D];   // 4 KB, contiguous block of hc
  const int tid  = threadIdx.x;
  const int row0 = blockIdx.x * 16;
  {
    // each thread async-copies 16 bytes: tile is 16 rows * 256 B, contiguous
    unsigned ldsAddr = (unsigned)(uintptr_t)(&Abf[tid * 8]);
    unsigned long long gAddr =
        (unsigned long long)(uintptr_t)(hcbf + (size_t)row0 * EMB_D + tid * 8);
    asm volatile("global_load_async_to_lds_b128 %0, %1, off"
                 :: "v"(ldsAddr), "v"(gAddr) : "memory");
    asm volatile("s_wait_asynccnt 0x0" ::: "memory");
  }
  __syncthreads();

  const int wave = tid >> 5, lane = tid & 31;
  const int half = lane >> 4, mr = lane & 15;
  const int nt   = blockIdx.y * 8 + wave;           // 0..319
  v8f acc = {0.f, 0.f, 0.f, 0.f, 0.f, 0.f, 0.f, 0.f};
#pragma unroll
  for (int kb = 0; kb < WO_KB; ++kb) {
    v8bf lo = *(const v8bf*)&Abf[mr * EMB_D + kb * 32 + half * 8];
    v8bf hi = *(const v8bf*)&Abf[mr * EMB_D + kb * 32 + 16 + half * 8];
    v16bf av = __builtin_shufflevector(lo, hi, 0, 1, 2, 3, 4, 5, 6, 7,
                                       8, 9, 10, 11, 12, 13, 14, 15);
    v16bf bv = *((const v16bf*)wofrag + ((size_t)(nt * WO_KB + kb) * 32 + lane));
    acc = __builtin_amdgcn_wmma_f32_16x16x32_bf16(false, av, false, bv, (short)0, acc, false, false);
  }

  int n = nt * 16 + mr;
  if (n < VOCAB_N) {
    float bb = bout[n];
#pragma unroll
    for (int r = 0; r < 8; ++r) {
      int m  = row0 + r + half * 8;    // m = t*512 + b
      int tt = m >> 9;
      int b  = m & 511;
      out[((size_t)b * L_STEPS + tt) * VOCAB_N + n] = acc[r] + bb;
    }
  }
}

// ---------------------------------------------------------------------------
extern "C" void kernel_launch(void* const* d_in, const int* in_sizes, int n_in,
                              void* d_out, int out_size, void* d_ws, size_t ws_size,
                              hipStream_t stream) {
  (void)in_sizes; (void)n_in; (void)out_size; (void)ws_size;
  const float* key    = (const float*)d_in[0];
  const float* values = (const float*)d_in[1];
  const float* mask   = (const float*)d_in[2];
  const int*   text   = (const int*)  d_in[3];
  const float* emb    = (const float*)d_in[4];
  const float* Wih1   = (const float*)d_in[5];
  const float* Whh1   = (const float*)d_in[6];
  const float* bih1   = (const float*)d_in[7];
  const float* bhh1   = (const float*)d_in[8];
  const float* Wih2   = (const float*)d_in[9];
  const float* Whh2   = (const float*)d_in[10];
  const float* bih2   = (const float*)d_in[11];
  const float* bhh2   = (const float*)d_in[12];
  const float* Wq     = (const float*)d_in[13];
  const float* bq     = (const float*)d_in[14];
  const float* Wout   = (const float*)d_in[15];
  const float* bout   = (const float*)d_in[16];
  float* ws = (float*)d_ws;

  dec_pack_kernel<<<1024, 256, 0, stream>>>(Wih1, Whh1, bih1, bhh1,
                                            Wih2, Whh2, bih2, bhh2,
                                            Wq, Wout, ws);

  dec_recurrent_kernel<<<B_SZ / 16, 256, 0, stream>>>(
      key, values, mask, text, emb, bq,
      ws + OFF_W1, ws + OFF_W2, ws + OFF_WQ, ws + OFF_B1, ws + OFF_B2,
      (__bf16*)(ws + OFF_HC));

  dim3 grid2((L_STEPS * B_SZ) / 16, WO_LDN / 128);
  dec_outgemm_kernel<<<grid2, 256, 0, stream>>>(
      (const __bf16*)(ws + OFF_HC), (const __bf16*)(ws + OFF_WO), bout, (float*)d_out);
}
